// similarity_computation_36197984370803
// MI455X (gfx1250) — compile-verified, matrix operand order checked by reading, not run on hardware
//
#include <hip/hip_runtime.h>
#include <hip/hip_bf16.h>

typedef __attribute__((ext_vector_type(16))) _Float16 v16h;
typedef __attribute__((ext_vector_type(8)))  _Float16 v8h;
typedef __attribute__((ext_vector_type(8)))  float    v8f;

// Problem constants
#define BB   4
#define DD   48
#define HH   64
#define WW   128
#define PADW 130            // W + 1-pixel zero halo each side
#define HWC  (HH * WW)      // 8192
#define CVOL 64             // 40 corr + 12 + 12
#define COUT 32
#define NGRP 40
#define CPG  8
#define NCC  12

// ---------------------------------------------------------------------------
// Stage 0a: fold BN params: scale = g*rsqrt(v+eps), shift = b - m*scale
// ---------------------------------------------------------------------------
__global__ void bn_fold_kernel(const float* __restrict__ g1, const float* __restrict__ b1,
                               const float* __restrict__ m1, const float* __restrict__ v1,
                               const float* __restrict__ g2, const float* __restrict__ b2,
                               const float* __restrict__ m2, const float* __restrict__ v2,
                               float* __restrict__ s1, float* __restrict__ sh1,
                               float* __restrict__ s2, float* __restrict__ sh2) {
    int i = threadIdx.x;
    if (i < 32) {
        float s = g1[i] * rsqrtf(v1[i] + 1e-5f);
        s1[i] = s;  sh1[i] = b1[i] - m1[i] * s;
    } else {
        int j = i - 32;
        float s = g2[j] * rsqrtf(v2[j] + 1e-5f);
        s2[j] = s;  sh2[j] = b2[j] - m2[j] * s;
    }
}

// ---------------------------------------------------------------------------
// Stage 0b: repack conv weights (Co, Ci, 3,3,3) f32 -> Bpt[n][k] f16 with
// k = tap*Cin + ci, n = co.  Row-per-n makes WMMA B-fragments 16B-contiguous.
// ---------------------------------------------------------------------------
__global__ void pack_weights_kernel(const float* __restrict__ w, _Float16* __restrict__ Bpt,
                                    int Cin, int total) {
    int idx = blockIdx.x * blockDim.x + threadIdx.x;
    if (idx >= total) return;
    int krow = 27 * Cin;
    int n   = idx / krow;
    int r   = idx % krow;
    int tap = r / Cin;
    int ci  = r % Cin;
    Bpt[idx] = (_Float16)w[(n * Cin + ci) * 27 + tap];
}

// ---------------------------------------------------------------------------
// Zero the W-halo columns (w'=0 and w'=129) of a channel-last tensor.
// ---------------------------------------------------------------------------
__global__ void zero_halo_kernel(_Float16* __restrict__ t, int C, int total) {
    int idx = blockIdx.x * blockDim.x + threadIdx.x;
    if (idx >= total) return;
    int ch  = idx % C;
    int col = (idx / C) & 1;
    int pix = idx / (2 * C);
    t[((size_t)pix * PADW + (col ? (PADW - 1) : 0)) * C + ch] = (_Float16)0.f;
}

// ---------------------------------------------------------------------------
// Stage 1: cost volume, f16, CHANNEL-LAST + W-halo: (B, D, H, 130, 64).
// Each thread assembles 64 channels in registers -> 8 contiguous b128 stores.
// Threads 0/1 also zero the halo columns.
// ---------------------------------------------------------------------------
__global__ __launch_bounds__(128)
void build_volume_kernel(const float* __restrict__ lg, const float* __restrict__ rg,
                         const float* __restrict__ lc, const float* __restrict__ rc,
                         _Float16* __restrict__ vol) {
    int blk = blockIdx.x;
    int h = blk & (HH - 1);
    int d = (blk >> 6) % DD;
    int b = blk / (DD * HH);
    int w = threadIdx.x;
    bool valid = (w >= d);
    int rw = valid ? (w - d) : 0;

    const float* lp  = lg + (size_t)b * 320 * HWC + (size_t)h * WW + w;
    const float* rp  = rg + (size_t)b * 320 * HWC + (size_t)h * WW + rw;
    const float* lcp = lc + (size_t)b * NCC * HWC + (size_t)h * WW + w;
    const float* rcp = rc + (size_t)b * NCC * HWC + (size_t)h * WW + rw;

    _Float16 o[CVOL];
#pragma unroll
    for (int g = 0; g < NGRP; ++g) {
        float acc = 0.f;
#pragma unroll
        for (int c = 0; c < CPG; ++c) {
            float lv = lp[(size_t)(g * CPG + c) * HWC];
            float rv = valid ? rp[(size_t)(g * CPG + c) * HWC] : 0.f;
            acc += lv * rv;
        }
        o[g] = (_Float16)(acc * 0.125f);
    }
#pragma unroll
    for (int j = 0; j < NCC; ++j) {
        o[NGRP + j]       = valid ? (_Float16)lcp[(size_t)j * HWC] : (_Float16)0.f;
        o[NGRP + NCC + j] = valid ? (_Float16)rcp[(size_t)j * HWC] : (_Float16)0.f;
    }

    _Float16* rowp = vol + (((size_t)b * DD + d) * HH + h) * ((size_t)PADW * CVOL);
    _Float16* vout = rowp + (size_t)(w + 1) * CVOL;
#pragma unroll
    for (int c8 = 0; c8 < 8; ++c8) {
        v8h t;
#pragma unroll
        for (int j = 0; j < 8; ++j) t[j] = o[c8 * 8 + j];
        *reinterpret_cast<v8h*>(vout + c8 * 8) = t;
    }
    if (w < 2) {  // zero halo columns 0 and 129
        _Float16* hp = rowp + (size_t)(w ? (PADW - 1) : 0) * CVOL;
        v8h z = {};
#pragma unroll
        for (int c8 = 0; c8 < 8; ++c8) *reinterpret_cast<v8h*>(hp + c8 * 8) = z;
    }
}

// ---------------------------------------------------------------------------
// Stage 2/3: LDS-free implicit-GEMM 3D conv via v_wmma_f32_16x16x32_f16.
// Channel-last + W-halo input: all in-plane loads are UNCONDITIONAL aligned
// b128; out-of-range (kd,kh) planes are skipped (their contribution is 0).
// 4 waves/WG, each computing a 32(M) x 32(N) register tile: per tap,
// 4 A-loads + 4 B-loads feed 4 WMMAs.  Epilogue folds BN + ReLU.
// ---------------------------------------------------------------------------
template <int CIN, bool OUT_F32>
__global__ __launch_bounds__(128)
void conv3d_wmma_kernel(const _Float16* __restrict__ in,    // (B, D, H, 130, CIN) f16
                        const _Float16* __restrict__ Bpt,   // (32, 27*CIN) f16 [n][k]
                        const float* __restrict__ scale,
                        const float* __restrict__ shift,
                        void* __restrict__ outv) {
    int blk = blockIdx.x;
    int h = blk & (HH - 1);
    int d = (blk >> 6) % DD;
    int b = blk / (DD * HH);
    int lane = threadIdx.x & 31;
    int wave = threadIdx.x >> 5;          // 0..3, covers M rows [wave*32, wave*32+32)

    // fragment roles (ISA 7.12.2, 16-bit A 16x32 / 32-bit C 16x16)
    const int mrow = (wave << 5) + (lane & 15);   // first M row of this lane
    const int kb   = (lane >> 4) << 3;            // lanes 16-31 start at K=8
    const int n0   = lane & 15;

    constexpr int CPT  = CIN / 32;                // K chunks per tap
    constexpr int KROW = 27 * CIN;

    const _Float16* bRow0 = Bpt + (size_t)n0 * KROW + kb;
    const _Float16* bRow1 = Bpt + (size_t)(n0 + 16) * KROW + kb;

    v8f c00 = {}, c01 = {}, c10 = {}, c11 = {};

    for (int cc = 0; cc < CPT; ++cc) {
        const int ci0 = cc * 32;
        for (int kd = 0; kd < 3; ++kd) {
            int id = d + kd - 1;
            if ((unsigned)id >= (unsigned)DD) continue;   // zero contribution
            for (int kh = 0; kh < 3; ++kh) {
                int ih = h + kh - 1;
                if ((unsigned)ih >= (unsigned)HH) continue;
                const _Float16* plane = in +
                    (((size_t)b * DD + id) * HH + ih) * ((size_t)PADW * CIN);
#pragma unroll
                for (int kw = 0; kw < 3; ++kw) {
                    int tap = (kd * 3 + kh) * 3 + kw;
                    // padded input column for output w=mrow is w' = mrow + kw
                    const _Float16* ap0 = plane + (size_t)(mrow + kw) * CIN + ci0 + kb;
                    const _Float16* ap1 = ap0 + 16 * CIN;
                    v8h a0lo = *reinterpret_cast<const v8h*>(ap0);
                    v8h a0hi = *reinterpret_cast<const v8h*>(ap0 + 16);
                    v8h a1lo = *reinterpret_cast<const v8h*>(ap1);
                    v8h a1hi = *reinterpret_cast<const v8h*>(ap1 + 16);

                    const _Float16* bp0 = bRow0 + tap * CIN + ci0;
                    const _Float16* bp1 = bRow1 + tap * CIN + ci0;
                    v8h b0lo = *reinterpret_cast<const v8h*>(bp0);
                    v8h b0hi = *reinterpret_cast<const v8h*>(bp0 + 16);
                    v8h b1lo = *reinterpret_cast<const v8h*>(bp1);
                    v8h b1hi = *reinterpret_cast<const v8h*>(bp1 + 16);

                    v16h a0, a1, bb0, bb1;
#pragma unroll
                    for (int i = 0; i < 8; ++i) {
                        a0[i] = a0lo[i];  a0[i + 8] = a0hi[i];
                        a1[i] = a1lo[i];  a1[i + 8] = a1hi[i];
                        bb0[i] = b0lo[i]; bb0[i + 8] = b0hi[i];
                        bb1[i] = b1lo[i]; bb1[i + 8] = b1hi[i];
                    }
                    c00 = __builtin_amdgcn_wmma_f32_16x16x32_f16(false, a0, false, bb0,
                                                                 (short)0, c00, false, false);
                    c01 = __builtin_amdgcn_wmma_f32_16x16x32_f16(false, a0, false, bb1,
                                                                 (short)0, c01, false, false);
                    c10 = __builtin_amdgcn_wmma_f32_16x16x32_f16(false, a1, false, bb0,
                                                                 (short)0, c10, false, false);
                    c11 = __builtin_amdgcn_wmma_f32_16x16x32_f16(false, a1, false, bb1,
                                                                 (short)0, c11, false, false);
                }
            }
        }
    }

    // ---- epilogue: BN + ReLU ----
    int wv0 = (wave << 5) + ((lane >> 4) << 3);    // M rows: +i (tile0), +16+i (tile1)
    float s0 = scale[n0],      t0 = shift[n0];
    float s1 = scale[n0 + 16], t1 = shift[n0 + 16];
    if constexpr (OUT_F32) {
        // final output: channel-first (B, 32, D, H, W) f32, contiguous along w
        size_t base0 = (((size_t)b * COUT + n0)      * DD + d) * HWC + (size_t)h * WW + wv0;
        size_t base1 = (((size_t)b * COUT + n0 + 16) * DD + d) * HWC + (size_t)h * WW + wv0;
        float* o = (float*)outv;
#pragma unroll
        for (int i = 0; i < 8; ++i) {
            o[base0 + i]      = fmaxf(c00[i] * s0 + t0, 0.f);
            o[base1 + i]      = fmaxf(c01[i] * s1 + t1, 0.f);
            o[base0 + 16 + i] = fmaxf(c10[i] * s0 + t0, 0.f);
            o[base1 + 16 + i] = fmaxf(c11[i] * s1 + t1, 0.f);
        }
    } else {
        // intermediate: channel-last (B, D, H, 130, 32) f16, halo-padded
        size_t pix = (((size_t)b * DD + d) * HH + h) * PADW + (wv0 + 1);
        _Float16* o = (_Float16*)outv + pix * COUT;
#pragma unroll
        for (int i = 0; i < 8; ++i) {
            o[(size_t)i * COUT + n0]               = (_Float16)fmaxf(c00[i] * s0 + t0, 0.f);
            o[(size_t)i * COUT + n0 + 16]          = (_Float16)fmaxf(c01[i] * s1 + t1, 0.f);
            o[(size_t)(i + 16) * COUT + n0]        = (_Float16)fmaxf(c10[i] * s0 + t0, 0.f);
            o[(size_t)(i + 16) * COUT + n0 + 16]   = (_Float16)fmaxf(c11[i] * s1 + t1, 0.f);
        }
    }
}

// ---------------------------------------------------------------------------
// Launch
// ---------------------------------------------------------------------------
extern "C" void kernel_launch(void* const* d_in, const int* in_sizes, int n_in,
                              void* d_out, int out_size, void* d_ws, size_t ws_size,
                              hipStream_t stream) {
    (void)in_sizes; (void)n_in; (void)out_size; (void)ws_size;

    const float* lg = (const float*)d_in[0];
    const float* rg = (const float*)d_in[1];
    const float* lc = (const float*)d_in[2];
    const float* rc = (const float*)d_in[3];
    const float* w1 = (const float*)d_in[4];
    const float* g1 = (const float*)d_in[5];
    const float* b1 = (const float*)d_in[6];
    const float* m1 = (const float*)d_in[7];
    const float* v1 = (const float*)d_in[8];
    const float* w2 = (const float*)d_in[9];
    const float* g2 = (const float*)d_in[10];
    const float* b2 = (const float*)d_in[11];
    const float* m2 = (const float*)d_in[12];
    const float* v2 = (const float*)d_in[13];

    // Workspace layout (bytes)
    const size_t NPIX      = (size_t)BB * DD * HH;               //  12,288 rows
    const size_t VOL_ELEMS = NPIX * PADW * CVOL;                 // 102,236,160
    const size_t X1_ELEMS  = NPIX * PADW * COUT;                 //  51,118,080
    const size_t BP1_ELEMS = 32ull * 27 * 64;                    //      55,296
    const size_t BP2_ELEMS = 32ull * 27 * 32;                    //      27,648

    char* ws = (char*)d_ws;
    _Float16* vol = (_Float16*)ws;
    _Float16* x1  = (_Float16*)(ws + VOL_ELEMS * 2);
    _Float16* Bp1 = (_Float16*)(ws + VOL_ELEMS * 2 + X1_ELEMS * 2);
    _Float16* Bp2 = (_Float16*)((char*)Bp1 + BP1_ELEMS * 2);
    float*    s1  = (float*)((char*)Bp2 + BP2_ELEMS * 2);
    float*    sh1 = s1 + 32;
    float*    s2  = s1 + 64;
    float*    sh2 = s1 + 96;

    // Stage 0: BN folding + weight repack + x1 halo zeroing
    bn_fold_kernel<<<1, 64, 0, stream>>>(g1, b1, m1, v1, g2, b2, m2, v2, s1, sh1, s2, sh2);
    pack_weights_kernel<<<(int)((BP1_ELEMS + 255) / 256), 256, 0, stream>>>(w1, Bp1, 64, (int)BP1_ELEMS);
    pack_weights_kernel<<<(int)((BP2_ELEMS + 255) / 256), 256, 0, stream>>>(w2, Bp2, 32, (int)BP2_ELEMS);
    {
        int total = (int)(NPIX * 2 * COUT);
        zero_halo_kernel<<<(total + 255) / 256, 256, 0, stream>>>(x1, COUT, total);
    }

    // Stage 1: channel-last f16 cost volume with W-halo
    build_volume_kernel<<<BB * DD * HH, 128, 0, stream>>>(lg, rg, lc, rc, vol);

    // Stage 2: conv1 + BN + ReLU -> channel-last f16 intermediate (halo-padded)
    conv3d_wmma_kernel<64, false><<<BB * DD * HH, 128, 0, stream>>>(vol, Bp1, s1, sh1, (void*)x1);

    // Stage 3: conv2 + BN + ReLU -> channel-first f32 output
    conv3d_wmma_kernel<32, true><<<BB * DD * HH, 128, 0, stream>>>(x1, Bp2, s2, sh2, d_out);
}
